// MGSTGNN_24945170055195
// MI455X (gfx1250) — compile-verified, matrix operand order checked by reading, not run on hardware
//
#include <hip/hip_runtime.h>

// ---------------- problem constants ----------------
#define BB   32     // batch
#define NN   307    // nodes
#define NP   320    // nodes padded to 16
#define NT16 20     // NP/16 tiles
#define EE   10     // embedding
#define HH   64     // hidden
#define CIN  65     // 1 + H
#define TT   12     // time steps
#define KP2  160    // padded K for xg @ W  (2*65 -> 160)
#define CP   80     // padded ias channel dim for L@X output tiles
#define NCT  3      // total GRU cells (1 + 2)

typedef __attribute__((ext_vector_type(16))) _Float16 v16h;
typedef __attribute__((ext_vector_type(8)))  float    v8f;

// ---- WMMA helpers (gfx1250 16x16x32 f16 -> f32) ----
// 16-bit A 16x32 layout: lane l holds row (l&15); halves[0..7] = K=kb..kb+7,
// halves[8..15] = K=16+kb..16+kb+7 with kb = 8*(l>>4). B is symmetric on a
// transposed (column-contiguous) buffer.
__device__ __forceinline__ v16h ld_frag(const _Float16* __restrict__ p) {
  v16h f;
#pragma unroll
  for (int j = 0; j < 8; ++j) { f[j] = p[j]; f[j + 8] = p[16 + j]; }
  return f;
}

__device__ __forceinline__ v8f wmma16(v16h a, v16h b, v8f c) {
  return __builtin_amdgcn_wmma_f32_16x16x32_f16(false, a, false, b, (short)0, c,
                                                false, false);
}

__device__ __forceinline__ float sigm(float x) { return 1.0f / (1.0f + expf(-x)); }

// ---------------- precompute kernels ----------------
// ne0[t,b,n,e] and x0[t,b,n]
__global__ void k_ne0(const float* __restrict__ src, const float* __restrict__ nemb,
                      const float* __restrict__ tid_e, const float* __restrict__ dow_e,
                      const float* __restrict__ hol_e, const float* __restrict__ hop_w,
                      const float* __restrict__ hop_b, float* __restrict__ ne0,
                      float* __restrict__ x0) {
  int idx = blockIdx.x * blockDim.x + threadIdx.x;
  if (idx >= BB * TT * NN) return;
  int n = idx % NN; int t = (idx / NN) % TT; int b = idx / (NN * TT);
  const float* s = src + ((size_t)(b * TT + t) * NN + n) * 5;
  int it = (int)(s[1] * 288.0f);
  int id = (int)s[2];
  int ih = (int)s[3];
  float hop = s[4];
  float* o = ne0 + ((size_t)(t * BB + b) * NN + n) * EE;
#pragma unroll
  for (int e = 0; e < EE; ++e) {
    float ne = nemb[n * EE + e] * tid_e[it * EE + e] * dow_e[id * EE + e] *
               hol_e[ih * EE + e];
    o[e] = ne * (hop * hop_w[e] + hop_b[e]);
  }
  x0[(size_t)(t * BB + b) * NN + n] = s[0];
}

// per-node weights: WT[cell][n][o][ki] = sum_e nemb[n,e]*wpool[cell,e,k,i,o], f16
// bT[cell][n][o] = sum_e nemb[n,e]*bpool[cell,e,o]
__global__ void k_prep_w(const float* __restrict__ wpool, const float* __restrict__ bpool,
                         const float* __restrict__ nemb, _Float16* __restrict__ WT,
                         float* __restrict__ bT, int Oout) {
  int idx = blockIdx.x * blockDim.x + threadIdx.x;
  if (idx >= NCT * NN * Oout) return;
  int o = idx % Oout; int n = (idx / Oout) % NN; int c = idx / (Oout * NN);
  float emb[EE];
#pragma unroll
  for (int e = 0; e < EE; ++e) emb[e] = nemb[n * EE + e];
  _Float16* w = WT + ((size_t)(c * NN + n) * Oout + o) * KP2;
  const size_t estr = (size_t)2 * CIN * Oout;
  for (int ki = 0; ki < KP2; ++ki) {
    float acc = 0.0f;
    if (ki < 2 * CIN) {
      int k = ki / CIN, i = ki % CIN;
      size_t base = (((size_t)c * EE * 2 + k) * CIN + i) * Oout + o;
#pragma unroll
      for (int e = 0; e < EE; ++e) acc += emb[e] * wpool[base + (size_t)e * estr];
    }
    w[ki] = (_Float16)acc;
  }
  float ab = 0.0f;
#pragma unroll
  for (int e = 0; e < EE; ++e) ab += emb[e] * bpool[(size_t)(c * EE + e) * Oout + o];
  bT[(size_t)(c * NN + n) * Oout + o] = ab;
}

// ---------------- per-step kernels ----------------
// builds ias^T (B,CP,NP) f16 (zero padded) and xg first half (node,b,k<65) f16
__global__ void k_build_ias(const float* __restrict__ x_t, const float* __restrict__ S,
                            _Float16* __restrict__ iasT, _Float16* __restrict__ xg) {
  int idx = blockIdx.x * blockDim.x + threadIdx.x;
  if (idx >= BB * NP) return;
  int n = idx % NP, b = idx / NP;
  float vals[CIN];
  if (n < NN) {
    vals[0] = x_t[(size_t)b * NN + n];
    const float* s = S + ((size_t)b * NN + n) * HH;
    for (int h = 0; h < HH; ++h) vals[1 + h] = s[h];
  } else {
    for (int i = 0; i < CIN; ++i) vals[i] = 0.0f;
  }
  _Float16* xp = xg + ((size_t)n * BB + b) * KP2;
  for (int i = 0; i < CIN; ++i) xp[i] = (_Float16)vals[i];
  _Float16* ip = iasT + (size_t)b * CP * NP + n;
  for (int c = 0; c < CP; ++c)
    ip[(size_t)c * NP] = (c < CIN) ? (_Float16)vals[c] : (_Float16)0.0f;
}

// tiny MLP + tanh(ne0*xe) -> nv16 (B,NP,32), zero padded in K and rows
__global__ void k_mlp_nv(const float* __restrict__ x_t, const float* __restrict__ S,
                         const float* __restrict__ ne0_t,
                         const float* __restrict__ w1, const float* __restrict__ b1,
                         const float* __restrict__ w2, const float* __restrict__ b2,
                         const float* __restrict__ w3, const float* __restrict__ b3,
                         _Float16* __restrict__ nv16) {
  int idx = blockIdx.x * blockDim.x + threadIdx.x;
  if (idx >= BB * NP) return;
  int n = idx % NP, b = idx / NP;
  _Float16* o = nv16 + ((size_t)b * NP + n) * 32;
  if (n >= NN) {
    for (int j = 0; j < 32; ++j) o[j] = (_Float16)0.0f;
    return;
  }
  float ias[CIN];
  ias[0] = x_t[(size_t)b * NN + n];
  const float* s = S + ((size_t)b * NN + n) * HH;
  for (int h = 0; h < HH; ++h) ias[1 + h] = s[h];
  float h1[16];
  for (int j = 0; j < 16; ++j) {
    float a = b1[j];
    for (int i = 0; i < CIN; ++i) a += ias[i] * w1[j * CIN + i];
    h1[j] = sigm(a);
  }
  float h2[2];
  for (int j = 0; j < 2; ++j) {
    float a = b2[j];
    for (int i = 0; i < 16; ++i) a += h1[i] * w2[j * 16 + i];
    h2[j] = sigm(a);
  }
  const float* ne0p = ne0_t + ((size_t)b * NN + n) * EE;
  for (int j = 0; j < 32; ++j) {
    float v = 0.0f;
    if (j < EE) {
      float xe = b3[j] + h2[0] * w3[j * 2 + 0] + h2[1] * w3[j * 2 + 1];
      v = tanhf(ne0p[j] * xe);
    }
    o[j] = (_Float16)v;
  }
}

// G = relu(NV @ NV^T), one 16x16 tile per wave, K=32 (E padded)
__global__ void k_wmma_g(const _Float16* __restrict__ nv16, float* __restrict__ g32) {
  int wave = (blockIdx.x * blockDim.x + threadIdx.x) >> 5;
  int lane = threadIdx.x & 31;
  if (wave >= BB * NT16 * NT16) return;
  int tj = wave % NT16; int ti = (wave / NT16) % NT16; int b = wave / (NT16 * NT16);
  int lr = lane & 15; int kb = (lane >> 4) * 8;
  const _Float16* base = nv16 + (size_t)b * NP * 32;
  v16h a  = ld_frag(base + (size_t)(ti * 16 + lr) * 32 + kb);
  v16h bf = ld_frag(base + (size_t)(tj * 16 + lr) * 32 + kb);  // B^T == NV (row-major)
  v8f c = {};
  c = wmma16(a, bf, c);
  float* o = g32 + ((size_t)b * NP + ti * 16) * NP + tj * 16;
  int mo = (lane >> 4) * 8;
#pragma unroll
  for (int r = 0; r < 8; ++r) o[(size_t)(mo + r) * NP + lr] = fmaxf(c[r], 0.0f);
}

__global__ void k_rowsum(const float* __restrict__ g32, float* __restrict__ d32) {
  int idx = blockIdx.x * blockDim.x + threadIdx.x;
  if (idx >= BB * NP) return;
  int n = idx % NP, b = idx / NP;
  const float4* r = (const float4*)(g32 + ((size_t)b * NP + n) * NP);
  float s = 0.0f;
  for (int m = 0; m < NP / 4; ++m) {
    float4 v = r[m];
    s += v.x + v.y + v.z + v.w;
  }
  d32[(size_t)b * NP + n] = (n < NN) ? (1.0f / sqrtf(s)) : 0.0f;
}

__global__ void k_scaleL(const float* __restrict__ g32, const float* __restrict__ d32,
                         _Float16* __restrict__ L16) {
  int idx = blockIdx.x * blockDim.x + threadIdx.x;
  if (idx >= BB * NP * NP / 4) return;
  size_t e0 = (size_t)idx * 4;
  int b = (int)(e0 / ((size_t)NP * NP));
  size_t rem = e0 % ((size_t)NP * NP);
  int n = (int)(rem / NP); int m0 = (int)(rem % NP);
  float dn = d32[(size_t)b * NP + n];
  const float* g = g32 + e0;
  const float* dm = d32 + (size_t)b * NP + m0;
  _Float16* o = L16 + e0;
#pragma unroll
  for (int j = 0; j < 4; ++j) o[j] = (_Float16)(g[j] * dn * dm[j]);
}

// Lx = L @ ias : one 16x16 output tile per wave, 10 K-steps; writes xg[*,*,65+c]
__global__ void k_wmma_lx(const _Float16* __restrict__ L16, const _Float16* __restrict__ iasT,
                          _Float16* __restrict__ xg) {
  int wave = (blockIdx.x * blockDim.x + threadIdx.x) >> 5;
  int lane = threadIdx.x & 31;
  if (wave >= BB * NT16 * (CP / 16)) return;
  int ct = wave % (CP / 16); int nt = (wave / (CP / 16)) % NT16;
  int b = wave / ((CP / 16) * NT16);
  int lr = lane & 15; int kb = (lane >> 4) * 8;
  const _Float16* Ab = L16 + ((size_t)b * NP + nt * 16 + lr) * NP + kb;
  const _Float16* Bb = iasT + ((size_t)b * CP + ct * 16 + lr) * NP + kb;
  v8f c = {};
#pragma unroll
  for (int kt = 0; kt < NP / 32; ++kt) {
    v16h a  = ld_frag(Ab + kt * 32);
    v16h bf = ld_frag(Bb + kt * 32);
    c = wmma16(a, bf, c);
  }
  int mo = (lane >> 4) * 8;
#pragma unroll
  for (int r = 0; r < 8; ++r) {
    int node = nt * 16 + mo + r;
    int cc = ct * 16 + lr;                       // 0..79; cols >=65 are zeros
    xg[((size_t)node * BB + b) * KP2 + CIN + cc] = (_Float16)c[r];
  }
}

// out[node,b,o] = xg[node] @ W[node] + bias : per-node (32 x 160) @ (160 x Oout)
__global__ void k_wmma_out(const _Float16* __restrict__ xg, const _Float16* __restrict__ WT,
                           const float* __restrict__ bT, float* __restrict__ gout,
                           int Oout, int OT) {
  int wave = (blockIdx.x * blockDim.x + threadIdx.x) >> 5;
  int lane = threadIdx.x & 31;
  if (wave >= NN * 2 * OT) return;
  int ot = wave % OT; int mt = (wave / OT) & 1; int node = wave / (OT * 2);
  int lr = lane & 15; int kb = (lane >> 4) * 8;
  const _Float16* Ab = xg + ((size_t)node * BB + mt * 16 + lr) * KP2 + kb;
  const _Float16* Bb = WT + ((size_t)node * Oout + ot * 16 + lr) * KP2 + kb;
  v8f c = {};
#pragma unroll
  for (int kt = 0; kt < KP2 / 32; ++kt) {
    v16h a  = ld_frag(Ab + kt * 32);
    v16h bf = ld_frag(Bb + kt * 32);
    c = wmma16(a, bf, c);
  }
  int mo = (lane >> 4) * 8;
#pragma unroll
  for (int r = 0; r < 8; ++r) {
    int brow = mt * 16 + mo + r;
    int o = ot * 16 + lr;
    gout[((size_t)node * BB + brow) * Oout + o] = c[r] + bT[(size_t)node * Oout + o];
  }
}

// ---------------- GRU elementwise ----------------
__global__ void k_zr(const float* __restrict__ gout_g, const float* __restrict__ S,
                     float* __restrict__ zs, float* __restrict__ r32) {
  int idx = blockIdx.x * blockDim.x + threadIdx.x;
  if (idx >= NN * BB * HH) return;
  int h = idx % HH; int b = (idx / HH) % BB; int n = idx / (HH * BB);
  const float* g = gout_g + ((size_t)n * BB + b) * (2 * HH);
  float z = sigm(g[h]);
  float r = sigm(g[HH + h]);
  size_t si = ((size_t)b * NN + n) * HH + h;
  zs[si] = z * S[si];
  r32[si] = r;
}

__global__ void k_hnew(const float* __restrict__ gout_u, float* __restrict__ S,
                       const float* __restrict__ r32, float* __restrict__ pred,
                       int writePred, int cslot, int koff) {
  int idx = blockIdx.x * blockDim.x + threadIdx.x;
  if (idx >= NN * BB * HH) return;
  int h = idx % HH; int b = (idx / HH) % BB; int n = idx / (HH * BB);
  float hc = tanhf(gout_u[((size_t)n * BB + b) * HH + h]);
  size_t si = ((size_t)b * NN + n) * HH + h;
  float r = r32[si];
  float hn = r * S[si] + (1.0f - r) * hc;
  S[si] = hn;
  if (writePred)
    pred[((size_t)(b * 2 + cslot) * NN + n) * (2 * HH) + koff + h] = hn;
}

// layer-1 input: x1[t,b,n] = skip_conv(layer0 last-2 outputs) + x0
__global__ void k_cur1(const float* __restrict__ pred, const float* __restrict__ skip_w,
                       const float* __restrict__ skip_b, const float* __restrict__ x0,
                       float* __restrict__ x1) {
  int idx = blockIdx.x * blockDim.x + threadIdx.x;
  if (idx >= TT * BB * NN) return;
  int n = idx % NN; int b = (idx / NN) % BB; int t = idx / (NN * BB);
  float acc = skip_b[t];
  for (int c = 0; c < 2; ++c) {
    const float* pr = pred + ((size_t)(b * 2 + c) * NN + n) * (2 * HH);  // layer0 half
    const float* sw = skip_w + (size_t)(t * 2 + c) * HH;
    for (int k = 0; k < HH; ++k) acc += pr[k] * sw[k];
  }
  size_t xi = (size_t)(t * BB + b) * NN + n;
  x1[xi] = acc + x0[xi];
}

__global__ void k_ln(const float* __restrict__ pred, const float* __restrict__ ln_g,
                     const float* __restrict__ ln_b, float* __restrict__ normed) {
  int idx = blockIdx.x * blockDim.x + threadIdx.x;
  if (idx >= BB * 2 * NN) return;
  const float* p = pred + (size_t)idx * (2 * HH);
  float* o = normed + (size_t)idx * (2 * HH);
  float m = 0.0f;
  for (int k = 0; k < 2 * HH; ++k) m += p[k];
  m *= (1.0f / (2 * HH));
  float v = 0.0f;
  for (int k = 0; k < 2 * HH; ++k) { float d = p[k] - m; v += d * d; }
  v *= (1.0f / (2 * HH));
  float inv = 1.0f / sqrtf(v + 1e-12f);
  for (int k = 0; k < 2 * HH; ++k) o[k] = (p[k] - m) * inv * ln_g[k] + ln_b[k];
}

__global__ void k_end(const float* __restrict__ normed, const float* __restrict__ end_w,
                      const float* __restrict__ end_b, float* __restrict__ out) {
  int idx = blockIdx.x * blockDim.x + threadIdx.x;
  if (idx >= BB * TT * NN) return;
  int n = idx % NN; int t = (idx / NN) % TT; int b = idx / (NN * TT);
  float acc = end_b[t];
  for (int c = 0; c < 2; ++c) {
    const float* nb = normed + ((size_t)(b * 2 + c) * NN + n) * (2 * HH);
    const float* ew = end_w + (size_t)(t * 2 + c) * (2 * HH);
    for (int k = 0; k < 2 * HH; ++k) acc += nb[k] * ew[k];
  }
  out[(size_t)(b * TT + t) * NN + n] = acc;
}

// ---------------- host orchestration ----------------
static inline int cdiv(int a, int b) { return (a + b - 1) / b; }

struct WSP {
  float *ne0, *x0, *x1, *bgT, *buT, *states, *g32, *d32, *gout_g, *gout_u, *zs, *r32,
      *pred, *normed;
  _Float16 *WgT, *WuT, *iasT, *nv16, *L16, *xg16;
};

static void run_gcn(const float* x_t, const float* Svar, const float* ne0_t, int cell,
                    const _Float16* WTbase, const float* bTbase, int Oout,
                    const float* w1, const float* b1, const float* w2, const float* b2,
                    const float* w3, const float* b3, float* gout, const WSP& P,
                    hipStream_t stream) {
  const int thr = 256;
  k_build_ias<<<cdiv(BB * NP, thr), thr, 0, stream>>>(x_t, Svar, P.iasT, P.xg16);
  k_mlp_nv<<<cdiv(BB * NP, thr), thr, 0, stream>>>(x_t, Svar, ne0_t, w1, b1, w2, b2, w3,
                                                   b3, P.nv16);
  k_wmma_g<<<cdiv(BB * NT16 * NT16 * 32, thr), thr, 0, stream>>>(P.nv16, P.g32);
  k_rowsum<<<cdiv(BB * NP, thr), thr, 0, stream>>>(P.g32, P.d32);
  k_scaleL<<<cdiv(BB * NP * NP / 4, thr), thr, 0, stream>>>(P.g32, P.d32, P.L16);
  k_wmma_lx<<<cdiv(BB * NT16 * (CP / 16) * 32, thr), thr, 0, stream>>>(P.L16, P.iasT,
                                                                       P.xg16);
  int OT = Oout / 16;
  k_wmma_out<<<cdiv(NN * 2 * OT * 32, thr), thr, 0, stream>>>(
      P.xg16, WTbase + (size_t)cell * NN * Oout * KP2, bTbase + (size_t)cell * NN * Oout,
      gout, Oout, OT);
}

extern "C" void kernel_launch(void* const* d_in, const int* in_sizes, int n_in,
                              void* d_out, int out_size, void* d_ws, size_t ws_size,
                              hipStream_t stream) {
  const float* src    = (const float*)d_in[0];
  const float* nemb   = (const float*)d_in[1];
  const float* tid_e  = (const float*)d_in[2];
  const float* dow_e  = (const float*)d_in[3];
  const float* hol_e  = (const float*)d_in[4];
  const float* hop_w  = (const float*)d_in[5];
  const float* hop_b  = (const float*)d_in[6];
  const float* gw     = (const float*)d_in[7];
  const float* gb     = (const float*)d_in[8];
  const float* g1w    = (const float*)d_in[9];
  const float* g1b    = (const float*)d_in[10];
  const float* g2w    = (const float*)d_in[11];
  const float* g2b    = (const float*)d_in[12];
  const float* g3w    = (const float*)d_in[13];
  const float* g3b    = (const float*)d_in[14];
  const float* uw     = (const float*)d_in[15];
  const float* ub     = (const float*)d_in[16];
  const float* u1w    = (const float*)d_in[17];
  const float* u1b    = (const float*)d_in[18];
  const float* u2w    = (const float*)d_in[19];
  const float* u2b    = (const float*)d_in[20];
  const float* u3w    = (const float*)d_in[21];
  const float* u3b    = (const float*)d_in[22];
  const float* skip_w = (const float*)d_in[23];
  const float* skip_b = (const float*)d_in[24];
  const float* ln_g   = (const float*)d_in[25];
  const float* ln_b   = (const float*)d_in[26];
  const float* end_w  = (const float*)d_in[27];
  const float* end_b  = (const float*)d_in[28];

  char* ws = (char*)d_ws;
  size_t off = 0;
  auto carve = [&](size_t bytes) -> void* {
    void* p = ws + off;
    off = (off + bytes + 255) & ~(size_t)255;
    return p;
  };

  WSP P;
  P.ne0    = (float*)carve((size_t)TT * BB * NN * EE * 4);
  P.x0     = (float*)carve((size_t)TT * BB * NN * 4);
  P.x1     = (float*)carve((size_t)TT * BB * NN * 4);
  P.WgT    = (_Float16*)carve((size_t)NCT * NN * 2 * HH * KP2 * 2);
  P.WuT    = (_Float16*)carve((size_t)NCT * NN * HH * KP2 * 2);
  P.bgT    = (float*)carve((size_t)NCT * NN * 2 * HH * 4);
  P.buT    = (float*)carve((size_t)NCT * NN * HH * 4);
  P.states = (float*)carve((size_t)2 * BB * NN * HH * 4);
  P.iasT   = (_Float16*)carve((size_t)BB * CP * NP * 2);
  P.nv16   = (_Float16*)carve((size_t)BB * NP * 32 * 2);
  P.g32    = (float*)carve((size_t)BB * NP * NP * 4);
  P.d32    = (float*)carve((size_t)BB * NP * 4);
  P.L16    = (_Float16*)carve((size_t)BB * NP * NP * 2);
  P.xg16   = (_Float16*)carve((size_t)NP * BB * KP2 * 2);
  P.gout_g = (float*)carve((size_t)NN * BB * 2 * HH * 4);
  P.gout_u = (float*)carve((size_t)NN * BB * HH * 4);
  P.zs     = (float*)carve((size_t)BB * NN * HH * 4);
  P.r32    = (float*)carve((size_t)BB * NN * HH * 4);
  P.pred   = (float*)carve((size_t)BB * 2 * NN * 2 * HH * 4);
  P.normed = (float*)carve((size_t)BB * 2 * NN * 2 * HH * 4);
  (void)ws_size;

  const int thr = 256;

  // zero xg pad region (cols >=145 never written elsewhere)
  hipMemsetAsync(P.xg16, 0, (size_t)NP * BB * KP2 * 2, stream);

  k_ne0<<<cdiv(BB * TT * NN, thr), thr, 0, stream>>>(src, nemb, tid_e, dow_e, hol_e,
                                                     hop_w, hop_b, P.ne0, P.x0);
  k_prep_w<<<cdiv(NCT * NN * 2 * HH, thr), thr, 0, stream>>>(gw, gb, nemb, P.WgT, P.bgT,
                                                             2 * HH);
  k_prep_w<<<cdiv(NCT * NN * HH, thr), thr, 0, stream>>>(uw, ub, nemb, P.WuT, P.buT, HH);

  for (int layer = 0; layer < 2; ++layer) {
    int ncell = (layer == 0) ? 1 : 2;
    int cell0 = (layer == 0) ? 0 : 1;
    const float* xbuf = (layer == 0) ? P.x0 : P.x1;
    hipMemsetAsync(P.states, 0, (size_t)ncell * BB * NN * HH * 4, stream);
    for (int t = 0; t < TT; ++t) {
      int idx = t % ncell;
      int cell = cell0 + idx;
      const float* x_t = xbuf + (size_t)t * BB * NN;
      float* S = P.states + (size_t)idx * BB * NN * HH;
      const float* ne0_t = P.ne0 + (size_t)t * BB * NN * EE;

      // gate GCN  -> zr
      run_gcn(x_t, S, ne0_t, cell, P.WgT, P.bgT, 2 * HH, g1w + cell * 16 * CIN,
              g1b + cell * 16, g2w + cell * 2 * 16, g2b + cell * 2, g3w + cell * EE * 2,
              g3b + cell * EE, P.gout_g, P, stream);
      k_zr<<<cdiv(NN * BB * HH, thr), thr, 0, stream>>>(P.gout_g, S, P.zs, P.r32);

      // candidate GCN on concat(x, z*state)
      run_gcn(x_t, P.zs, ne0_t, cell, P.WuT, P.buT, HH, u1w + cell * 16 * CIN,
              u1b + cell * 16, u2w + cell * 2 * 16, u2b + cell * 2, u3w + cell * EE * 2,
              u3b + cell * EE, P.gout_u, P, stream);
      int wp = (t >= TT - 2) ? 1 : 0;
      k_hnew<<<cdiv(NN * BB * HH, thr), thr, 0, stream>>>(P.gout_u, S, P.r32, P.pred, wp,
                                                          t - (TT - 2), layer * HH);
    }
    if (layer == 0)
      k_cur1<<<cdiv(TT * BB * NN, thr), thr, 0, stream>>>(P.pred, skip_w, skip_b, P.x0,
                                                          P.x1);
  }

  k_ln<<<cdiv(BB * 2 * NN, thr), thr, 0, stream>>>(P.pred, ln_g, ln_b, P.normed);
  k_end<<<cdiv(BB * TT * NN, thr), thr, 0, stream>>>(P.normed, end_w, end_b,
                                                     (float*)d_out);
}
